// DualTerrainPrior_33483565040116
// MI455X (gfx1250) — compile-verified, matrix operand order checked by reading, not run on hardware
//
#include <hip/hip_runtime.h>

typedef __attribute__((ext_vector_type(16))) _Float16 v16h;
typedef __attribute__((ext_vector_type(8)))  float    v8f;

#define G 160
#define BETA_L 0.02f
#define BETA_S 0.05f
#define SL 68   // LDS row stride (floats) for 64-wide rows; 68*4=272 bytes (16B aligned)
#define SS 20   // LDS row stride (floats) for 16-wide rows; 80 bytes (16B aligned)
#define WAVES 2
#define BLOCK (WAVES * 32)

// K index held in half 'h' of a v16h A/B fragment for V_WMMA_*_16X16X32 (16-bit),
// per CDNA5 ISA 7.12.2: lanes 0-15 hold K 0..7 (VGPR0-3) and 16..23 (VGPR4-7),
// lanes 16-31 hold K 8..15 and 24..31.
__device__ __forceinline__ int kmap(int lane, int h) {
    int hi = (lane & 16) ? 8 : 0;
    return (h < 8) ? (h + hi) : (h + 8 + hi);
}

__device__ __forceinline__ float sigmoidf_(float x) {
    return 1.0f / (1.0f + __expf(-x));
}

// Async DMA: global -> LDS, 16 bytes per lane, tracked on ASYNCcnt.
// LDS byte offset = low 32 bits of the flat shared-memory pointer
// (LDS aperture truncation, CDNA5 ISA 10.2).
__device__ __forceinline__ void async_g2l_b128(const float* gaddr, float* ldsPtr) {
    unsigned lds_off = (unsigned)(uintptr_t)ldsPtr;
    unsigned long long ga = (unsigned long long)(uintptr_t)gaddr;
    asm volatile("global_load_async_to_lds_b128 %0, %1, off"
                 :: "v"(lds_off), "v"(ga)
                 : "memory");
}

__device__ __forceinline__ void wait_async_all() {
    asm volatile("s_wait_asynccnt 0x0" ::: "memory");
}

__device__ __forceinline__ void trisample(const float* __restrict__ grid,
                                          const float* z, float* out5) {
    int   c0[3];
    float f[3];
#pragma unroll
    for (int j = 0; j < 3; ++j) {
        float c = (z[j] + 1.0f) * 0.5f * (float)(G - 1);
        int   fl = (int)floorf(c);
        fl = fl < 0 ? 0 : (fl > G - 2 ? G - 2 : fl);
        c0[j] = fl;
        f[j]  = c - (float)fl;
    }
#pragma unroll
    for (int j = 0; j < 5; ++j) out5[j] = 0.0f;
#pragma unroll
    for (int dx = 0; dx < 2; ++dx) {
        float wx = dx ? f[0] : 1.0f - f[0];
#pragma unroll
        for (int dy = 0; dy < 2; ++dy) {
            float wy = dy ? f[1] : 1.0f - f[1];
#pragma unroll
            for (int dz = 0; dz < 2; ++dz) {
                float wz = dz ? f[2] : 1.0f - f[2];
                float w  = wx * wy * wz;
                const float* p = grid +
                    (((c0[0] + dx) * G + (c0[1] + dy)) * G + (c0[2] + dz)) * 5;
#pragma unroll
                for (int j = 0; j < 5; ++j) out5[j] += w * p[j];
            }
        }
    }
}

__global__ __launch_bounds__(BLOCK) void dual_terrain_kernel(
    const float* __restrict__ q_l,  const float* __restrict__ q_s,
    const float* __restrict__ ter_l,const float* __restrict__ ter_s,
    const float* __restrict__ Wt_l, const float* __restrict__ bt_l,
    const float* __restrict__ Wpq_l,const float* __restrict__ a_h_l,
    const float* __restrict__ a_e_l,const float* __restrict__ b_g_l,
    const float* __restrict__ Wt_s, const float* __restrict__ bt_s,
    const float* __restrict__ Wpq_s,const float* __restrict__ a_h_s,
    const float* __restrict__ a_e_s,const float* __restrict__ b_g_s,
    const float* __restrict__ w_l,  const float* __restrict__ w_s,
    float* __restrict__ out_qtl, float* __restrict__ out_qts,
    float* __restrict__ out_g,   float* __restrict__ out_sig)
{
    __shared__ float qbl[WAVES][32 * SL];  // q_ltm tile, token-major
    __shared__ float qbs[WAVES][32 * SS];  // q_stm tile
    __shared__ float dbl[WAVES][32 * SL];  // delta_q_ltm
    __shared__ float dbs[WAVES][32 * SS];  // delta_q_stm
    __shared__ float zbl[WAVES][32 * 4];   // z_ltm (pre-tanh), later inv-norm
    __shared__ float zbs[WAVES][32 * 4];
    __shared__ float sgl[WAVES][32 * 8];   // sig_ltm (5 used)
    __shared__ float sgs[WAVES][32 * 8];

    const int tid  = threadIdx.x;
    const int lane = tid & 31;
    const int wv   = tid >> 5;
    const int t0   = blockIdx.x * (WAVES * 32) + wv * 32;  // first token of wave
    const int n    = lane & 15;

    // ---- kick off async DMA of q tiles into LDS (no VGPR round-trip) ----
    {
        const float* srcL = q_l + (size_t)t0 * 64;
#pragma unroll
        for (int j = 0; j < 16; ++j) {
            int e4 = lane + 32 * j;          // float4 index inside 2048-float tile
            int tm = e4 >> 4;
            int k  = (e4 & 15) << 2;
            async_g2l_b128(srcL + e4 * 4, &qbl[wv][tm * SL + k]);
        }
        const float* srcS = q_s + (size_t)t0 * 16;
#pragma unroll
        for (int j = 0; j < 4; ++j) {
            int e4 = lane + 32 * j;
            int tm = e4 >> 2;
            int k  = (e4 & 3) << 2;
            async_g2l_b128(srcS + e4 * 4, &qbs[wv][tm * SS + k]);
        }
    }

    // ---- B fragments (weights) built while the DMA is in flight ----
    v16h bzl0, bzl1, bzs, bdl0, bdl1, bdl2, bdl3, bds;
#pragma unroll
    for (int h = 0; h < 16; ++h) {
        int k = kmap(lane, h);
        _Float16 z0 = (_Float16)0.0f;
        bzl0[h] = (n < 3) ? (_Float16)Wt_l[n * 64 + k]      : z0;   // B[k][n]=Wt[n][k]
        bzl1[h] = (n < 3) ? (_Float16)Wt_l[n * 64 + 32 + k] : z0;
        bzs[h]  = (n < 3 && k < 16) ? (_Float16)Wt_s[n * 16 + k] : z0;
        bdl0[h] = (k < 5) ? (_Float16)Wpq_l[( 0 + n) * 5 + k] : z0; // B[k][n]=Wpq[n'][k]
        bdl1[h] = (k < 5) ? (_Float16)Wpq_l[(16 + n) * 5 + k] : z0;
        bdl2[h] = (k < 5) ? (_Float16)Wpq_l[(32 + n) * 5 + k] : z0;
        bdl3[h] = (k < 5) ? (_Float16)Wpq_l[(48 + n) * 5 + k] : z0;
        bds[h]  = (k < 5) ? (_Float16)Wpq_s[n * 5 + k] : z0;
    }

    wait_async_all();
    __syncthreads();

    // ---- z = q . Wt^T via WMMA (f16 operands, f32 accumulate) ----
#pragma unroll
    for (int tt = 0; tt < 2; ++tt) {
        int m = tt * 16 + n;                 // A-row owned by this lane
        v16h a0, a1, as_;
#pragma unroll
        for (int h = 0; h < 16; ++h) {
            int k = kmap(lane, h);
            a0[h]  = (_Float16)qbl[wv][m * SL + k];
            a1[h]  = (_Float16)qbl[wv][m * SL + 32 + k];
            as_[h] = (k < 16) ? (_Float16)qbs[wv][m * SS + k] : (_Float16)0.0f;
        }
        v8f cz = {};
        cz = __builtin_amdgcn_wmma_f32_16x16x32_f16(false, a0, false, bzl0,
                                                    (short)0, cz, false, false);
        cz = __builtin_amdgcn_wmma_f32_16x16x32_f16(false, a1, false, bzl1,
                                                    (short)0, cz, false, false);
        v8f cs = {};
        cs = __builtin_amdgcn_wmma_f32_16x16x32_f16(false, as_, false, bzs,
                                                    (short)0, cs, false, false);
        if (n < 3) {                         // only 3 output columns are real
            int mb = tt * 16 + ((lane >> 4) << 3);
#pragma unroll
            for (int r = 0; r < 8; ++r) {
                zbl[wv][(mb + r) * 4 + n] = cz[r];
                zbs[wv][(mb + r) * 4 + n] = cs[r];
            }
        }
    }
    __syncthreads();

    // ---- per-token: tanh, trilinear gather (L2-resident grids), gate, signal ----
    const int tok = t0 + lane;
    float zl[3], zs2[3];
#pragma unroll
    for (int j = 0; j < 3; ++j) {
        zl[j]  = tanhf(zbl[wv][lane * 4 + j] + bt_l[j]);
        zs2[j] = tanhf(zbs[wv][lane * 4 + j] + bt_s[j]);
    }
    float sl5[5], ss5[5];
    trisample(ter_l, zl,  sl5);
    trisample(ter_s, zs2, ss5);
#pragma unroll
    for (int j = 0; j < 5; ++j) {
        sgl[wv][lane * 8 + j] = sl5[j];
        sgs[wv][lane * 8 + j] = ss5[j];
        out_sig[(size_t)tok * 10 + j]     = sl5[j];
        out_sig[(size_t)tok * 10 + 5 + j] = ss5[j];
    }
    {
        float gl = sigmoidf_(a_h_l[0] * sl5[0] + a_e_l[0] * sl5[1] + a_e_l[1] * sl5[2] +
                             a_e_l[2] * sl5[3] + a_e_l[3] * sl5[4] + b_g_l[0]);
        float gs = sigmoidf_(a_h_s[0] * ss5[0] + a_e_s[0] * ss5[1] + a_e_s[1] * ss5[2] +
                             a_e_s[2] * ss5[3] + a_e_s[3] * ss5[4] + b_g_s[0]);
        float lw = sigmoidf_(w_l[0]);
        float sw = sigmoidf_(w_s[0]);
        out_g[tok] = (lw * gl + sw * gs) / (lw + sw);
    }
    __syncthreads();

    // ---- delta_q = sig . Wpq^T via WMMA ----
#pragma unroll
    for (int tt = 0; tt < 2; ++tt) {
        int m = tt * 16 + n;
        v16h al, as_;
#pragma unroll
        for (int h = 0; h < 16; ++h) {
            int k = kmap(lane, h);
            al[h]  = (k < 5) ? (_Float16)sgl[wv][m * 8 + k] : (_Float16)0.0f;
            as_[h] = (k < 5) ? (_Float16)sgs[wv][m * 8 + k] : (_Float16)0.0f;
        }
        v8f d0 = {}, d1 = {}, d2 = {}, d3 = {}, dv = {};
        d0 = __builtin_amdgcn_wmma_f32_16x16x32_f16(false, al, false, bdl0,
                                                    (short)0, d0, false, false);
        d1 = __builtin_amdgcn_wmma_f32_16x16x32_f16(false, al, false, bdl1,
                                                    (short)0, d1, false, false);
        d2 = __builtin_amdgcn_wmma_f32_16x16x32_f16(false, al, false, bdl2,
                                                    (short)0, d2, false, false);
        d3 = __builtin_amdgcn_wmma_f32_16x16x32_f16(false, al, false, bdl3,
                                                    (short)0, d3, false, false);
        dv = __builtin_amdgcn_wmma_f32_16x16x32_f16(false, as_, false, bds,
                                                    (short)0, dv, false, false);
        int mb = tt * 16 + ((lane >> 4) << 3);
#pragma unroll
        for (int r = 0; r < 8; ++r) {
            dbl[wv][(mb + r) * SL +      n] = d0[r];
            dbl[wv][(mb + r) * SL + 16 + n] = d1[r];
            dbl[wv][(mb + r) * SL + 32 + n] = d2[r];
            dbl[wv][(mb + r) * SL + 48 + n] = d3[r];
            dbs[wv][(mb + r) * SS +      n] = dv[r];
        }
    }
    __syncthreads();

    // ---- per-token inverse norms (reuse zbl/zbs as broadcast buffers) ----
    {
        float sql = 0.0f, sqs = 0.0f;
#pragma unroll
        for (int d = 0; d < 64; ++d) {
            float v = qbl[wv][lane * SL + d] + BETA_L * dbl[wv][lane * SL + d];
            sql += v * v;
        }
#pragma unroll
        for (int d = 0; d < 16; ++d) {
            float v = qbs[wv][lane * SS + d] + BETA_S * dbs[wv][lane * SS + d];
            sqs += v * v;
        }
        float nl = sqrtf(sql); nl = nl > 1e-12f ? nl : 1e-12f;
        float ns = sqrtf(sqs); ns = ns > 1e-12f ? ns : 1e-12f;
        zbl[wv][lane] = 1.0f / nl;
        zbs[wv][lane] = 1.0f / ns;
    }
    __syncthreads();

    // ---- coalesced normalized outputs ----
    {
        float4* dst = (float4*)(out_qtl + (size_t)t0 * 64);
#pragma unroll
        for (int j = 0; j < 16; ++j) {
            int e4 = lane + 32 * j;
            int tm = e4 >> 4;
            int k  = (e4 & 15) << 2;
            float  inv = zbl[wv][tm];
            float4 q4  = *(float4*)&qbl[wv][tm * SL + k];
            float4 d4  = *(float4*)&dbl[wv][tm * SL + k];
            float4 o;
            o.x = (q4.x + BETA_L * d4.x) * inv;
            o.y = (q4.y + BETA_L * d4.y) * inv;
            o.z = (q4.z + BETA_L * d4.z) * inv;
            o.w = (q4.w + BETA_L * d4.w) * inv;
            dst[e4] = o;
        }
        float4* dss = (float4*)(out_qts + (size_t)t0 * 16);
#pragma unroll
        for (int j = 0; j < 4; ++j) {
            int e4 = lane + 32 * j;
            int tm = e4 >> 2;
            int k  = (e4 & 3) << 2;
            float  inv = zbs[wv][tm];
            float4 q4  = *(float4*)&qbs[wv][tm * SS + k];
            float4 d4  = *(float4*)&dbs[wv][tm * SS + k];
            float4 o;
            o.x = (q4.x + BETA_S * d4.x) * inv;
            o.y = (q4.y + BETA_S * d4.y) * inv;
            o.z = (q4.z + BETA_S * d4.z) * inv;
            o.w = (q4.w + BETA_S * d4.w) * inv;
            dss[e4] = o;
        }
    }
}

extern "C" void kernel_launch(void* const* d_in, const int* in_sizes, int n_in,
                              void* d_out, int out_size, void* d_ws, size_t ws_size,
                              hipStream_t stream) {
    (void)in_sizes; (void)n_in; (void)out_size; (void)d_ws; (void)ws_size;
    const float* q_l   = (const float*)d_in[0];
    const float* q_s   = (const float*)d_in[1];
    const float* ter_l = (const float*)d_in[2];
    const float* ter_s = (const float*)d_in[3];
    const float* Wt_l  = (const float*)d_in[4];
    const float* bt_l  = (const float*)d_in[5];
    const float* Wpq_l = (const float*)d_in[6];
    const float* a_h_l = (const float*)d_in[7];
    const float* a_e_l = (const float*)d_in[8];
    const float* b_g_l = (const float*)d_in[9];
    const float* Wt_s  = (const float*)d_in[10];
    const float* bt_s  = (const float*)d_in[11];
    const float* Wpq_s = (const float*)d_in[12];
    const float* a_h_s = (const float*)d_in[13];
    const float* a_e_s = (const float*)d_in[14];
    const float* b_g_s = (const float*)d_in[15];
    const float* w_l   = (const float*)d_in[16];
    const float* w_s   = (const float*)d_in[17];

    const size_t BT = (size_t)16 * 16384;
    float* out     = (float*)d_out;
    float* out_qtl = out;                 // [B,T,64]
    float* out_qts = out_qtl + BT * 64;   // [B,T,16]
    float* out_g   = out_qts + BT * 16;   // [B,T,1]
    float* out_sig = out_g + BT;          // [B,T,10]

    dim3 grid((unsigned)(BT / BLOCK));
    dim3 block(BLOCK);
    hipLaunchKernelGGL(dual_terrain_kernel, grid, block, 0, stream,
                       q_l, q_s, ter_l, ter_s,
                       Wt_l, bt_l, Wpq_l, a_h_l, a_e_l, b_g_l,
                       Wt_s, bt_s, Wpq_s, a_h_s, a_e_s, b_g_s,
                       w_l, w_s,
                       out_qtl, out_qts, out_g, out_sig);
}
